// MemoryContrastiveLossPRISM_80805514707393
// MI455X (gfx1250) — matrix operand order repaired and checked by reading, not compiled, not run on hardware
//
#include <hip/hip_runtime.h>
#include <hip/hip_bf16.h>
#include <math.h>

typedef __attribute__((ext_vector_type(16))) __bf16 v16bf;
typedef __attribute__((ext_vector_type(8)))  __bf16 v8bf;
typedef __attribute__((ext_vector_type(4)))  __bf16 v4bf;
typedef __attribute__((ext_vector_type(8)))  float  v8f;
typedef __attribute__((ext_vector_type(4)))  float  f32x4;

#define MARGIN 0.5f
#define POSTH  (1.0f - 1e-5f)

// ---------------------------------------------------------------------------
// Kernel 1: per-row online softmax over class logits, pick prob at target.
// ---------------------------------------------------------------------------
__global__ __launch_bounds__(256)
void prism_softmax_sel(const float* __restrict__ Acol,
                       const float* __restrict__ center,
                       const int*   __restrict__ targets,
                       const unsigned char* __restrict__ filled,
                       float* __restrict__ Cout,
                       int d, int ncls) {
    __shared__ float row[512];
    __shared__ float redM[256];
    __shared__ float redS[256];
    __shared__ float ztarget;

    const int i = blockIdx.x;
    const int t = threadIdx.x;

    for (int k = t; k < d; k += blockDim.x) row[k] = Acol[(size_t)i * d + k];
    const int tgt = targets[i];
    __syncthreads();

    float m = -3.4e38f, s = 0.0f;
    for (int c = t; c < ncls; c += blockDim.x) {
        const float* cp = center + (size_t)c * d;
        float z = 0.0f;
        #pragma unroll 8
        for (int k = 0; k < d; ++k) z = fmaf(row[k], cp[k], z);
        if (c == tgt) ztarget = z;        // unique writer
        if (z > m) { s *= __expf(m - z); m = z; }
        s += __expf(z - m);
    }
    redM[t] = m; redS[t] = s;
    __syncthreads();
    for (int off = 128; off > 0; off >>= 1) {
        if (t < off) {
            float m1 = redM[t], s1 = redS[t];
            float m2 = redM[t + off], s2 = redS[t + off];
            float M = fmaxf(m1, m2);
            redS[t] = s1 * __expf(m1 - M) + s2 * __expf(m2 - M);
            redM[t] = M;
        }
        __syncthreads();
    }
    if (t == 0) {
        float sel = __expf(ztarget - redM[0]) / redS[0];
        Cout[i] = filled[tgt] ? sel : 1.0f;
    }
}

// ---------------------------------------------------------------------------
// Kernel 2: keep[i] = (ascending rank of C[i] with index tiebreak) >= n/2.
// ---------------------------------------------------------------------------
__global__ __launch_bounds__(256)
void rank_keep(const float* __restrict__ C, int n, int nRemove,
               int* __restrict__ keepI, float* __restrict__ out) {
    const int i = blockIdx.x * blockDim.x + threadIdx.x;
    if (i >= n) return;
    const float ci = C[i];
    int cnt = 0;
    for (int j = 0; j < n; ++j) {
        float cj = C[j];
        cnt += (cj < ci) || (cj == ci && j < i);
    }
    const int kp = (cnt >= nRemove) ? 1 : 0;
    keepI[i]   = kp;
    out[1 + i] = (float)kp;               // keep mask as float output
}

// ---------------------------------------------------------------------------
// Kernel P: one-shot fp32 -> bf16 prepack (bandwidth bound).
// Removes per-tile re-conversion and halves GEMM operand traffic.
// ---------------------------------------------------------------------------
__global__ __launch_bounds__(256)
void prepack_bf16(const float* __restrict__ src, __bf16* __restrict__ dst,
                  long long count4) {           // count/4 work items
    const long long i = (long long)blockIdx.x * blockDim.x + threadIdx.x;
    if (i >= count4) return;
    f32x4 x = *(const f32x4*)(src + i * 4);
    v4bf  y;
    #pragma unroll
    for (int e = 0; e < 4; ++e) y[e] = (__bf16)x[e];
    *(v4bf*)(dst + i * 4) = y;
}

// ---------------------------------------------------------------------------
// Fragment loaders
// ---------------------------------------------------------------------------
__device__ __forceinline__ v16bf load_a_frag_bf(const __bf16* __restrict__ p) {
    v8bf x0 = *(const v8bf*)(p);        // K {0..7}   (lane-group base folded in)
    v8bf x1 = *(const v8bf*)(p + 8);    // K {8..15}
    v16bf a;
    #pragma unroll
    for (int e = 0; e < 8; ++e) { a[e] = x0[e]; a[8 + e] = x1[e]; }
    return a;
}

__device__ __forceinline__ v16bf load_b_frag_bf(const __bf16* __restrict__ p) {
    v8bf x0 = *(const v8bf*)(p);        // 16 contiguous K values
    v8bf x1 = *(const v8bf*)(p + 8);
    v16bf b;
    #pragma unroll
    for (int e = 0; e < 8; ++e) { b[e] = x0[e]; b[8 + e] = x1[e]; }
    return b;
}

__device__ __forceinline__ v16bf load_a_frag_f32(const float* __restrict__ p) {
    f32x4 x0 = *(const f32x4*)(p);
    f32x4 x1 = *(const f32x4*)(p + 4);
    f32x4 x2 = *(const f32x4*)(p + 8);
    f32x4 x3 = *(const f32x4*)(p + 12);
    v16bf a;
    #pragma unroll
    for (int e = 0; e < 4; ++e) {
        a[e] = (__bf16)x0[e]; a[4 + e] = (__bf16)x1[e];
        a[8 + e] = (__bf16)x2[e]; a[12 + e] = (__bf16)x3[e];
    }
    return a;
}

__device__ __forceinline__ v16bf load_b_frag_f32(const float* __restrict__ p) {
    f32x4 x0 = *(const f32x4*)(p);
    f32x4 x1 = *(const f32x4*)(p + 4);
    f32x4 x2 = *(const f32x4*)(p + 8);
    f32x4 x3 = *(const f32x4*)(p + 12);
    v16bf b;
    #pragma unroll
    for (int e = 0; e < 4; ++e) {
        b[e] = (__bf16)x0[e]; b[4 + e] = (__bf16)x1[e];
        b[8 + e] = (__bf16)x2[e]; b[12 + e] = (__bf16)x3[e];
    }
    return b;
}

__device__ __forceinline__ v8f wmma_bf16(v16bf a, v16bf b, v8f c) {
    return __builtin_amdgcn_wmma_f32_16x16x32_bf16(
        false, a, false, b, (short)0, c, false, false);
}

// ---------------------------------------------------------------------------
// Shared epilogue: branchless masked reduce of a 32x32 macro-tile + block sum.
// ---------------------------------------------------------------------------
__device__ __forceinline__ void tile_epilogue(
        v8f c00, v8f c01, v8f c10, v8f c11,
        const int* sTc, const float* sKeep, float* sWave,
        int trj0, int trj1, int lane, int wave,
        float* __restrict__ partials) {
    const int mbase = (lane < 16) ? 0 : 8;
    auto tile_acc = [&](const v8f& cc, int rowOff, int trj) -> float {
        float a = 0.0f;
        #pragma unroll
        for (int r = 0; r < 8; ++r) {
            const int mrow = rowOff + mbase + r;
            const float s = cc[r];
            const float posC = (s < POSTH)  ? (1.0f - s) : 0.0f;
            const float negC = (s > MARGIN) ? s          : 0.0f;
            const float sel  = (sTc[mrow] == trj) ? posC : negC;
            a = fmaf(sKeep[mrow], sel, a);
        }
        return a;
    };
    float acc = tile_acc(c00, 0,  trj0) + tile_acc(c01, 0,  trj1)
              + tile_acc(c10, 16, trj0) + tile_acc(c11, 16, trj1);

    #pragma unroll
    for (int off = 16; off > 0; off >>= 1)
        acc += __shfl_xor(acc, off, 32);
    if (lane == 0) sWave[wave] = acc;
    __syncthreads();

    if (threadIdx.x == 0) {
        float bsum = 0.0f;
        #pragma unroll
        for (int w = 0; w < 8; ++w) bsum += sWave[w];
        partials[blockIdx.y * gridDim.x + blockIdx.x] = bsum;
    }
}

// ---------------------------------------------------------------------------
// Kernel 3a: fused GEMM on prepacked bf16 operands. Hot loop = pure b128
// loads + 4 WMMAs per k-step. 32x32 macro-tile/wave, 8 waves/block.
// ---------------------------------------------------------------------------
__global__ __launch_bounds__(256)
void loss_wmma_bf16(const __bf16* __restrict__ A,   // [n, d] bf16
                    const __bf16* __restrict__ R,   // [m, d] bf16
                    const int*   __restrict__ tcol,
                    const int*   __restrict__ trow,
                    const int*   __restrict__ keepI,
                    float* __restrict__ partials,
                    int d) {
    __shared__ int   sTc[32];
    __shared__ float sKeep[32];
    __shared__ float sWave[8];

    const int lane = threadIdx.x & 31;
    const int wave = threadIdx.x >> 5;
    const int rowBase = blockIdx.y << 5;
    const int colBase = (blockIdx.x << 8) + (wave << 5);
    const int l15 = lane & 15;

    if (threadIdx.x < 32) {
        sTc[threadIdx.x]   = tcol[rowBase + threadIdx.x];
        sKeep[threadIdx.x] = (float)keepI[rowBase + threadIdx.x];
    }

    const int akoff = (lane < 16) ? 0 : 16;
    const __bf16* aptr0 = A + (size_t)(rowBase + l15) * d + akoff;
    const __bf16* aptr1 = aptr0 + (size_t)16 * d;
    const int bkoff = (lane < 16) ? 0 : 16;
    const __bf16* bptr0 = R + (size_t)(colBase + l15) * d + bkoff;
    const __bf16* bptr1 = bptr0 + (size_t)16 * d;

    const int trj0 = trow[colBase + l15];
    const int trj1 = trow[colBase + 16 + l15];

    v8f c00 = {}, c01 = {}, c10 = {}, c11 = {};
    #pragma unroll 2
    for (int kk = 0; kk < d; kk += 32) {
        v16bf a0 = load_a_frag_bf(aptr0 + kk);
        v16bf a1 = load_a_frag_bf(aptr1 + kk);
        v16bf b0 = load_b_frag_bf(bptr0 + kk);
        v16bf b1 = load_b_frag_bf(bptr1 + kk);
        c00 = wmma_bf16(a0, b0, c00);
        c01 = wmma_bf16(a0, b1, c01);
        c10 = wmma_bf16(a1, b0, c10);
        c11 = wmma_bf16(a1, b1, c11);
    }

    __syncthreads();
    tile_epilogue(c00, c01, c10, c11, sTc, sKeep, sWave,
                  trj0, trj1, lane, wave, partials);
}

// ---------------------------------------------------------------------------
// Kernel 3b: fallback GEMM reading fp32 + converting in-loop (small ws).
// ---------------------------------------------------------------------------
__global__ __launch_bounds__(256)
void loss_wmma_f32(const float* __restrict__ A,
                   const float* __restrict__ R,
                   const int*   __restrict__ tcol,
                   const int*   __restrict__ trow,
                   const int*   __restrict__ keepI,
                   float* __restrict__ partials,
                   int d) {
    __shared__ int   sTc[32];
    __shared__ float sKeep[32];
    __shared__ float sWave[8];

    const int lane = threadIdx.x & 31;
    const int wave = threadIdx.x >> 5;
    const int rowBase = blockIdx.y << 5;
    const int colBase = (blockIdx.x << 8) + (wave << 5);
    const int l15 = lane & 15;

    if (threadIdx.x < 32) {
        sTc[threadIdx.x]   = tcol[rowBase + threadIdx.x];
        sKeep[threadIdx.x] = (float)keepI[rowBase + threadIdx.x];
    }

    const int akoff = (lane < 16) ? 0 : 16;
    const float* aptr0 = A + (size_t)(rowBase + l15) * d + akoff;
    const float* aptr1 = aptr0 + (size_t)16 * d;
    const int bkoff = (lane < 16) ? 0 : 16;
    const float* bptr0 = R + (size_t)(colBase + l15) * d + bkoff;
    const float* bptr1 = bptr0 + (size_t)16 * d;

    const int trj0 = trow[colBase + l15];
    const int trj1 = trow[colBase + 16 + l15];

    v8f c00 = {}, c01 = {}, c10 = {}, c11 = {};
    #pragma unroll 2
    for (int kk = 0; kk < d; kk += 32) {
        v16bf a0 = load_a_frag_f32(aptr0 + kk);
        v16bf a1 = load_a_frag_f32(aptr1 + kk);
        v16bf b0 = load_b_frag_f32(bptr0 + kk);
        v16bf b1 = load_b_frag_f32(bptr1 + kk);
        c00 = wmma_bf16(a0, b0, c00);
        c01 = wmma_bf16(a0, b1, c01);
        c10 = wmma_bf16(a1, b0, c10);
        c11 = wmma_bf16(a1, b1, c11);
    }

    __syncthreads();
    tile_epilogue(c00, c01, c10, c11, sTc, sKeep, sWave,
                  trj0, trj1, lane, wave, partials);
}

// ---------------------------------------------------------------------------
// Kernel 4: tree-reduce the per-block partials into out[0] = sum / n.
// ---------------------------------------------------------------------------
__global__ __launch_bounds__(256)
void final_reduce(const float* __restrict__ partials, int np, float invN,
                  float* __restrict__ out) {
    __shared__ float red[256];
    float s = 0.0f;
    for (int i = threadIdx.x; i < np; i += 256) s += partials[i];
    red[threadIdx.x] = s;
    __syncthreads();
    for (int off = 128; off > 0; off >>= 1) {
        if (threadIdx.x < off) red[threadIdx.x] += red[threadIdx.x + off];
        __syncthreads();
    }
    if (threadIdx.x == 0) out[0] = red[0] * invN;
}

// ---------------------------------------------------------------------------
extern "C" void kernel_launch(void* const* d_in, const int* in_sizes, int n_in,
                              void* d_out, int out_size, void* d_ws, size_t ws_size,
                              hipStream_t stream) {
    const float*         Acol   = (const float*)d_in[0];         // inputs_col [n,d]
    const float*         Brow   = (const float*)d_in[1];         // inputs_row [m,d]
    const float*         center = (const float*)d_in[2];         // [c,d]
    const int*           tcol   = (const int*)d_in[3];           // [n]
    const int*           trow   = (const int*)d_in[4];           // [m]
    const unsigned char* filled = (const unsigned char*)d_in[5]; // [c]

    const int n    = in_sizes[3];          // 1024
    const int m    = in_sizes[4];          // 65536
    const int d    = in_sizes[0] / n;      // 512
    const int ncls = in_sizes[5];          // 10000

    float* out = (float*)d_out;            // [0]=loss, [1..n]=keep

    // Workspace layout (16B-aligned sections):
    //   Cvals[n] f32 | keepI[n] i32 | parts[np] f32 | Abf[n*d] bf16 | Rbf[m*d] bf16
    const int np = (m / 256) * (n / 32);
    char* ws = (char*)d_ws;
    float* Cvals = (float*)ws;
    int*   keepI = (int*)(ws + (size_t)n * 4);
    float* parts = (float*)(ws + (size_t)n * 8);
    size_t bfOff = ((size_t)n * 8 + (size_t)np * 4 + 15) & ~(size_t)15;
    __bf16* Abf  = (__bf16*)(ws + bfOff);
    __bf16* Rbf  = Abf + (size_t)n * d;
    const size_t needBytes = bfOff + ((size_t)n + (size_t)m) * d * 2;
    const bool usePrepack = (ws_size >= needBytes);

    // 1) per-row softmax prob at target class
    prism_softmax_sel<<<n, 256, 0, stream>>>(Acol, center, tcol, filled,
                                             Cvals, d, ncls);
    // 2) rank-based keep mask
    rank_keep<<<(n + 255) / 256, 256, 0, stream>>>(Cvals, n, n / 2, keepI, out);

    // 3) fused WMMA GEMM + masked contrastive reduction -> per-block partials
    dim3 grid(m / 256, n / 32);
    if (usePrepack) {
        const long long cntA4 = ((long long)n * d) / 4;
        const long long cntR4 = ((long long)m * d) / 4;
        prepack_bf16<<<(int)((cntA4 + 255) / 256), 256, 0, stream>>>(Acol, Abf, cntA4);
        prepack_bf16<<<(int)((cntR4 + 255) / 256), 256, 0, stream>>>(Brow, Rbf, cntR4);
        loss_wmma_bf16<<<grid, 256, 0, stream>>>(Abf, Rbf, tcol, trow, keepI,
                                                 parts, d);
    } else {
        loss_wmma_f32<<<grid, 256, 0, stream>>>(Acol, Brow, tcol, trow, keepI,
                                                parts, d);
    }

    // 4) reduce partials -> out[0]
    final_reduce<<<1, 256, 0, stream>>>(parts, np, 1.0f / (float)n, out);
}
